// GDRCore_48077863912185
// MI455X (gfx1250) — matrix operand order repaired and checked by reading, not compiled, not run on hardware
//
#include <hip/hip_runtime.h>
#include <hip/hip_bf16.h>

// ---------------------------------------------------------------------------
// Gated delta-rule erase-write memory update for MI455X (gfx1250, wave32).
// Two HBM-streaming GEMMs done with v_wmma_f32_16x16x32_bf16 (f32 accumulate,
// bf16 operands converted in registers -> memory-bound at 23.3 TB/s).
// Inner GEMM loops are software-pipelined (ping-pong fragment buffers) so
// global_load_b128 batches for chunk j+1 are in flight while chunk j's WMMAs
// execute, avoiding full s_wait_loadcnt 0 stalls before every WMMA.
// ---------------------------------------------------------------------------

typedef __attribute__((ext_vector_type(16))) __bf16 v16bf;
typedef __attribute__((ext_vector_type(8)))  float  v8f;

#define HW    4096
#define KDIM  64
#define CDIM  256
#define BDIM  2
#define NDIM  16
#define NH    8
#define BN    (BDIM * NDIM)

union BF16x16 { v16bf v; uint4 u[2]; };

static __device__ __forceinline__ v8f wmma_bf16(v16bf a, v16bf b, v8f c) {
  // (neg_a, A, neg_b, B, c_mod, C, reuse_a, reuse_b)
  return __builtin_amdgcn_wmma_f32_16x16x32_bf16(false, a, false, b,
                                                 (short)0, c, false, false);
}

static __device__ __forceinline__ void cvt4(v16bf& d, int o, float4 f) {
  d[o + 0] = (__bf16)f.x; d[o + 1] = (__bf16)f.y;
  d[o + 2] = (__bf16)f.z; d[o + 3] = (__bf16)f.w;
}

// -------------------------------------------------------------------------
// K1: softmax over K (64) per (b, hw). Writes key_n as bf16 in both
// [b,k,hw] (row major) and [b,hw,k] (transposed) layouts for WMMA feeding.
// -------------------------------------------------------------------------
__global__ __launch_bounds__(256) void k_softmax(const float* __restrict__ kmap,
                                                 __bf16* __restrict__ keybf,
                                                 __bf16* __restrict__ knt) {
  int idx = blockIdx.x * 256 + threadIdx.x;      // 0 .. B*HW-1 (8192)
  int b  = idx >> 12;
  int hw = idx & (HW - 1);
  const float* src = kmap + (size_t)b * KDIM * HW + hw;

  float v[KDIM];
  float m = -3.0e38f;
#pragma unroll
  for (int k = 0; k < KDIM; ++k) { v[k] = src[(size_t)k * HW]; m = fmaxf(m, v[k]); }
  float s = 0.f;
#pragma unroll
  for (int k = 0; k < KDIM; ++k) { v[k] = __expf(v[k] - m); s += v[k]; }
  float inv = 1.f / s;

  __bf16 row[KDIM];
#pragma unroll
  for (int k = 0; k < KDIM; ++k) {
    __bf16 pb = (__bf16)(v[k] * inv);
    keybf[((size_t)b * KDIM + k) * HW + hw] = pb;
    row[k] = pb;
  }
  uint4* dst = (uint4*)(knt + ((size_t)b * HW + hw) * KDIM);
  const uint4* sr = (const uint4*)row;
#pragma unroll
  for (int i = 0; i < 8; ++i) dst[i] = sr[i];
}

// -------------------------------------------------------------------------
// K2: gram[b,k,j] = sum_hw key_n[b,k,hw]*key_n[b,j,hw]. WMMA split-K with
// f32 atomic accumulate. 256 waves: b(2) x jtile(4) x hw-chunk(32 of 128).
// -------------------------------------------------------------------------
__global__ __launch_bounds__(128) void k_gram(const __bf16* __restrict__ keybf,
                                              float* __restrict__ gram) {
  int w    = blockIdx.x * 4 + (threadIdx.x >> 5);
  int lane = threadIdx.x & 31;
  int chunk = w & 31;
  int jtile = (w >> 5) & 3;
  int b     = w >> 7;
  int  lm = lane & 15;
  bool lo = lane < 16;
  const __bf16* base = keybf + (size_t)b * KDIM * HW;

  v8f acc[4] = {};
  int hws = chunk * 128;
  for (int hw0 = hws; hw0 < hws + 128; hw0 += 32) {
    BF16x16 bf;
    {
      const __bf16* p = base + (size_t)(jtile * 16 + lm) * HW + hw0 + (lo ? 0 : 8);
      bf.u[0] = *(const uint4*)p;
      bf.u[1] = *(const uint4*)(p + 16);
    }
#pragma unroll
    for (int t = 0; t < 4; ++t) {
      BF16x16 af;
      const __bf16* p = base + (size_t)(t * 16 + lm) * HW + hw0 + (lo ? 0 : 8);
      af.u[0] = *(const uint4*)p;
      af.u[1] = *(const uint4*)(p + 16);
      acc[t] = wmma_bf16(af.v, bf.v, acc[t]);
    }
  }
#pragma unroll
  for (int t = 0; t < 4; ++t)
#pragma unroll
    for (int r = 0; r < 8; ++r) {
      int k = t * 16 + r + (lo ? 0 : 8);
      int j = jtile * 16 + lm;
      atomicAdd(gram + ((size_t)b * KDIM + k) * KDIM + j, acc[t][r]);
    }
}

// -------------------------------------------------------------------------
// K3: vk_t[b,n,k,v] = sum_hw key_n[b,k,hw]*value[b,n,v,hw].
// Per wave: 64(k) x 16(v) strip, value f32 loaded once & converted to bf16
// in registers. HW split 8-way (4096 waves), f32 atomic accumulate.
// Software-pipelined: chunk j+1 fragments load while chunk j WMMAs execute.
// -------------------------------------------------------------------------
__global__ __launch_bounds__(128) void k_vkt(const float* __restrict__ value,
                                             const __bf16* __restrict__ keybf,
                                             float* __restrict__ vkt) {
  int w    = blockIdx.x * 4 + (threadIdx.x >> 5);
  int lane = threadIdx.x & 31;
  int chunk = w & 7;
  int vtile = (w >> 3) & 15;
  int bn    = w >> 7;
  int b     = bn >> 4;
  int  lm = lane & 15;
  bool lo = lane < 16;

  // Per-lane base pointers (row parts folded in once).
  const __bf16* ap = keybf + (size_t)b * KDIM * HW + (size_t)lm * HW + (lo ? 0 : 8);
  const float*  vp = value + ((size_t)bn * CDIM + vtile * 16 + lm) * HW + (lo ? 0 : 8);

  auto LOADA = [&](int hw0, BF16x16* a) {
#pragma unroll
    for (int t = 0; t < 4; ++t) {
      const __bf16* p = ap + (size_t)t * 16 * HW + hw0;
      a[t].u[0] = *(const uint4*)p;
      a[t].u[1] = *(const uint4*)(p + 16);
    }
  };
  auto LOADB = [&](int hw0, float4* r) {
    const float4* p = (const float4*)(vp + hw0);
    r[0] = p[0]; r[1] = p[1]; r[2] = p[4]; r[3] = p[5];
  };

  v8f acc[4] = {};
  int hws = chunk * 512;

  BF16x16 abuf[2][4];
  float4  rbuf[2][4];
  LOADA(hws, abuf[0]);
  LOADB(hws, rbuf[0]);
#pragma unroll
  for (int j = 0; j < 16; ++j) {
    int cur = j & 1, nxt = cur ^ 1;
    if (j < 15) {                       // prefetch chunk j+1
      LOADA(hws + (j + 1) * 32, abuf[nxt]);
      LOADB(hws + (j + 1) * 32, rbuf[nxt]);
    }
    BF16x16 bfr;                        // convert at consume time only
    cvt4(bfr.v, 0,  rbuf[cur][0]);
    cvt4(bfr.v, 4,  rbuf[cur][1]);
    cvt4(bfr.v, 8,  rbuf[cur][2]);
    cvt4(bfr.v, 12, rbuf[cur][3]);
#pragma unroll
    for (int t = 0; t < 4; ++t)
      acc[t] = wmma_bf16(abuf[cur][t].v, bfr.v, acc[t]);
  }

#pragma unroll
  for (int t = 0; t < 4; ++t)
#pragma unroll
    for (int r = 0; r < 8; ++r) {
      int k = t * 16 + r + (lo ? 0 : 8);
      int v = vtile * 16 + lm;
      atomicAdd(vkt + ((size_t)bn * KDIM + k) * CDIM + v, acc[t][r]);
    }
}

// -------------------------------------------------------------------------
// K4: per-(b,n) workgroup. Stages state (64KB) in LDS; computes beta/alpha
// gates, v_k = gram @ state, the erase-write update, and emits new_state
// transposed as bf16 [b,n,v,k] via an LDS transpose buffer.
// -------------------------------------------------------------------------
__global__ __launch_bounds__(256) void k_update(const float* __restrict__ state,
                                                const float* __restrict__ bw,
                                                const float* __restrict__ aw,
                                                const float* __restrict__ Alog,
                                                const float* __restrict__ dtb,
                                                const float* __restrict__ gram,
                                                const float* __restrict__ vkt,
                                                __bf16* __restrict__ nst) {
  extern __shared__ char smem[];
  float*  s_state = (float*)smem;                        // 64*256 f32 = 64KB
  __bf16* s_nst   = (__bf16*)(smem + 65536);             // 256*64 bf16 = 32KB
  float*  s_beta  = (float*)(smem + 65536 + 32768);      // 512 f32
  float*  s_alpha = s_beta + KDIM * NH;                  // 512 f32

  int bn = blockIdx.x;
  int b  = bn >> 4;
  int t  = threadIdx.x;
  const float* st = state + (size_t)bn * KDIM * CDIM;

  for (int i = t; i < KDIM * CDIM / 4; i += 256)
    ((float4*)s_state)[i] = ((const float4*)st)[i];
  __syncthreads();

  for (int p = t; p < KDIM * NH; p += 256) {
    int k = p >> 3, h = p & 7;
    const float* srow = s_state + k * CDIM;
    const float* bwr  = bw + h * CDIM;
    const float* awr  = aw + h * CDIM;
    float db = 0.f, da = 0.f;
#pragma unroll 8
    for (int c = 0; c < CDIM; ++c) { float s = srow[c]; db += s * bwr[c]; da += s * awr[c]; }
    float beta = 1.f / (1.f + __expf(-db));
    float x  = da + dtb[h];
    float sp = (x > 20.f) ? x : log1pf(__expf(x));
    s_beta[p]  = beta;
    s_alpha[p] = -__expf(Alog[h]) * sp;
  }
  __syncthreads();

  const float* gbase  = gram + (size_t)b * KDIM * KDIM;
  const float* vk_row = vkt + (size_t)bn * KDIM * CDIM;
  int v = t;            // 0..255, this thread owns column v for all k
  int h = v >> 5;       // rep = C/NUM_HEADS = 32
  for (int k = 0; k < KDIM; ++k) {
    const float* g = gbase + k * KDIM;
    float vkv = 0.f;
#pragma unroll 8
    for (int j = 0; j < KDIM; ++j) vkv += g[j] * s_state[j * CDIM + v];
    float beta  = s_beta[k * NH + h];
    float alpha = s_alpha[k * NH + h];
    float sv = s_state[k * CDIM + v];
    float wt = vk_row[k * CDIM + v];
    float ns = alpha * (sv - beta * vkv) + beta * wt;
    s_nst[v * KDIM + k] = (__bf16)ns;
  }
  __syncthreads();

  __bf16* out = nst + (size_t)bn * CDIM * KDIM;
  for (int i = t; i < CDIM * KDIM / 2; i += 256)
    ((unsigned int*)out)[i] = ((const unsigned int*)s_nst)[i];
}

// -------------------------------------------------------------------------
// K5: readout[b,n,v,hw] = sum_k key_n[b,k,hw]*new_state[b,n,k,v].
// M=v(256), N=hw(4096), Kred=64 -> 2 WMMAs per 16x16 tile; A fragments
// hoisted per wave; B fragments double-buffered across the 8 hw-tiles;
// coalesced f32 stores of the 134MB output.
// 16384 waves: bn(32) x vtile(16) x hwblock(32 of 8 tiles).
// -------------------------------------------------------------------------
__global__ __launch_bounds__(128) void k_readout(const __bf16* __restrict__ nst,
                                                 const __bf16* __restrict__ knt,
                                                 float* __restrict__ out) {
  int w    = blockIdx.x * 4 + (threadIdx.x >> 5);
  int lane = threadIdx.x & 31;
  int hwblk = w & 31;
  int vtile = (w >> 5) & 15;
  int bn    = w >> 9;
  int b     = bn >> 4;
  int  lm = lane & 15;
  bool lo = lane < 16;

  BF16x16 a0, a1;                       // hoisted A fragments (kk=0, kk=32)
  {
    const __bf16* p  = nst + ((size_t)bn * CDIM + vtile * 16 + lm) * KDIM;
    const __bf16* p0 = p + (lo ? 0 : 8);
    a0.u[0] = *(const uint4*)p0;        a0.u[1] = *(const uint4*)(p0 + 16);
    const __bf16* p1 = p0 + 32;
    a1.u[0] = *(const uint4*)p1;        a1.u[1] = *(const uint4*)(p1 + 16);
  }

  const __bf16* kb = knt + (size_t)b * HW * KDIM + (size_t)lm * KDIM + (lo ? 0 : 8);
  float* ob = out + ((size_t)bn * CDIM + vtile * 16) * HW;
  int ht0 = hwblk * 8;

  auto LOADB = [&](int ht, BF16x16& b0, BF16x16& b1) {
    const __bf16* p = kb + (size_t)ht * 16 * KDIM;
    b0.u[0] = *(const uint4*)p;          b0.u[1] = *(const uint4*)(p + 16);
    b1.u[0] = *(const uint4*)(p + 32);   b1.u[1] = *(const uint4*)(p + 48);
  };

  BF16x16 b0buf[2], b1buf[2];
  LOADB(ht0, b0buf[0], b1buf[0]);
#pragma unroll
  for (int j = 0; j < 8; ++j) {
    int cur = j & 1, nxt = cur ^ 1;
    if (j < 7) LOADB(ht0 + j + 1, b0buf[nxt], b1buf[nxt]);   // prefetch

    v8f acc = {};
    acc = wmma_bf16(a0.v, b0buf[cur].v, acc);
    acc = wmma_bf16(a1.v, b1buf[cur].v, acc);

    int ht = ht0 + j;
#pragma unroll
    for (int r = 0; r < 8; ++r) {
      int m = r + (lo ? 0 : 8);
      ob[(size_t)m * HW + ht * 16 + lm] = acc[r];
    }
  }
}

// -------------------------------------------------------------------------
extern "C" void kernel_launch(void* const* d_in, const int* in_sizes, int n_in,
                              void* d_out, int out_size, void* d_ws, size_t ws_size,
                              hipStream_t stream) {
  (void)in_sizes; (void)n_in; (void)out_size; (void)ws_size;
  const float* value = (const float*)d_in[0];
  const float* kmap  = (const float*)d_in[1];
  const float* state = (const float*)d_in[2];
  const float* bw    = (const float*)d_in[3];
  const float* aw    = (const float*)d_in[4];
  const float* Alog  = (const float*)d_in[5];
  const float* dtb   = (const float*)d_in[6];
  float* out = (float*)d_out;

  char* ws = (char*)d_ws;
  __bf16* keybf = (__bf16*)(ws);                      // 1 MB  [B,K,HW] bf16
  __bf16* knt   = (__bf16*)(ws + (1u << 20));         // 1 MB  [B,HW,K] bf16
  float*  gram  = (float*)(ws + (2u << 20));          // 32 KB [B,K,K]  f32
  float*  vkt   = (float*)(ws + (3u << 20));          // 2 MB  [B,N,K,C] f32
  __bf16* nst   = (__bf16*)(ws + (11u << 20));        // 1 MB  [B,N,C,K] bf16

  hipMemsetAsync(gram, 0, (size_t)BDIM * KDIM * KDIM * sizeof(float), stream);
  hipMemsetAsync(vkt, 0, (size_t)BN * KDIM * CDIM * sizeof(float), stream);

  k_softmax<<<BDIM * HW / 256, 256, 0, stream>>>(kmap, keybf, knt);
  k_gram<<<64, 128, 0, stream>>>(keybf, gram);
  k_vkt<<<1024, 128, 0, stream>>>(value, keybf, vkt);
  k_update<<<BN, 256, 102400, stream>>>(state, bw, aw, Alog, dtb, gram, vkt, nst);
  k_readout<<<4096, 128, 0, stream>>>(nst, knt, out);
}
